// CausalSelfAttention_17179869226
// MI455X (gfx1250) — matrix operand order repaired and checked by reading, not compiled, not run on hardware
//
#include <hip/hip_runtime.h>
#include <hip/hip_bf16.h>
#include <math.h>

// ---------------------------------------------------------------------------
// Types
// ---------------------------------------------------------------------------
typedef __bf16 bf16_t;
typedef __attribute__((ext_vector_type(16))) __bf16 v16bf;
typedef __attribute__((ext_vector_type(8)))  __bf16 v8bf;
typedef __attribute__((ext_vector_type(8)))  float  v8f;

union V16U { v16bf v; v8bf h[2]; };

// Problem constants (match reference)
constexpr int CB = 4;          // batch
constexpr int CS = 2048;       // sequence
constexpr int CE = 2048;       // embed
constexpr int CH = 16;         // heads
constexpr int CD = 128;        // head dim
constexpr int CM = CB * CS;    // 8192 rows

__device__ __forceinline__ bf16_t f2bf(float f) {
    unsigned u = __builtin_bit_cast(unsigned, f);
    u += 0x7FFFu + ((u >> 16) & 1u);            // round-to-nearest-even
    unsigned short h = (unsigned short)(u >> 16);
    return __builtin_bit_cast(bf16_t, h);
}
__device__ __forceinline__ float bf2f(bf16_t b) {
    unsigned short h = __builtin_bit_cast(unsigned short, b);
    unsigned u = ((unsigned)h) << 16;
    return __builtin_bit_cast(float, u);
}

__device__ __forceinline__ v8f wmma_bf16(v16bf a, v16bf b, v8f c) {
    return __builtin_amdgcn_wmma_f32_16x16x32_bf16(
        /*neg_a=*/false, a, /*neg_b=*/false, b,
        /*c_mod=*/(short)0, c, /*reuse_a=*/false, /*reuse_b=*/false);
}

__device__ __forceinline__ v8f v8f_zero() {
    v8f z;
#pragma unroll
    for (int j = 0; j < 8; ++j) z[j] = 0.0f;
    return z;
}

// 32B async DMA global -> LDS (two b128, inst offset applies to both sides)
__device__ __forceinline__ void async_copy32(const bf16_t* g, bf16_t* l) {
    const unsigned lds = (unsigned)(size_t)(void*)l;          // addr[31:0] = LDS byte offset
    const unsigned long long ga = (unsigned long long)(size_t)g;
    asm volatile(
        "global_load_async_to_lds_b128 %0, %1, off\n\t"
        "global_load_async_to_lds_b128 %0, %1, off offset:16"
        :: "v"(lds), "v"(ga) : "memory");
}

// ---------------------------------------------------------------------------
// 1) f32 -> bf16 elementwise convert (x)
// ---------------------------------------------------------------------------
__global__ void cvt_f32_bf16_kernel(const float* __restrict__ in,
                                    bf16_t* __restrict__ out, long long n) {
    long long i = ((long long)blockIdx.x * blockDim.x + threadIdx.x) * 4;
    if (i + 3 < n) {
        float4 f = *(const float4*)(in + i);
        out[i + 0] = f2bf(f.x);
        out[i + 1] = f2bf(f.y);
        out[i + 2] = f2bf(f.z);
        out[i + 3] = f2bf(f.w);
    }
}

// ---------------------------------------------------------------------------
// 2) f32 [K,N] -> bf16 [N,K] transpose-convert (weights)
// ---------------------------------------------------------------------------
__global__ void transpose_cvt_kernel(const float* __restrict__ w,
                                     bf16_t* __restrict__ wt, int K, int N) {
    long long id = (long long)blockIdx.x * blockDim.x + threadIdx.x;
    long long total = (long long)K * N;
    if (id < total) {
        int k = (int)(id / N);
        int n = (int)(id % N);
        wt[(long long)n * K + k] = f2bf(w[id]);
    }
}

// ---------------------------------------------------------------------------
// 3/6) bf16 WMMA GEMM: C[M,N] = A[M,K] * W[K,N], W given transposed as Bt[N,K]
//      block = 256 threads = 8 waves (2 along M, 4 along N)
//      block tile 64(M) x 256(N), wave tile 32 x 64, K step 32
// ---------------------------------------------------------------------------
template <bool OUT_F32>
__global__ __launch_bounds__(256, 1) void gemm_bf16_kernel(
        const bf16_t* __restrict__ A,
        const bf16_t* __restrict__ Bt,
        void* __restrict__ Cout,
        int M, int N, int K) {
    const int lane = threadIdx.x & 31;
    const int wave = threadIdx.x >> 5;
    const int li = lane & 15;
    const int hi = lane >> 4;
    const int mw = wave & 1;
    const int nw = wave >> 1;
    const int m0 = blockIdx.x * 64 + mw * 32;
    const int n0 = blockIdx.y * 256 + nw * 64;

    v8f acc[2][4];
#pragma unroll
    for (int r = 0; r < 2; ++r)
#pragma unroll
        for (int t = 0; t < 4; ++t) acc[r][t] = v8f_zero();

    const bf16_t* arow0 = A + (long long)(m0 + li) * K;
    const bf16_t* arow1 = A + (long long)(m0 + 16 + li) * K;

#pragma unroll 1
    for (int k0 = 0; k0 < K; k0 += 32) {
        V16U a0, a1;
        a0.h[0] = *(const v8bf*)(arow0 + k0 + hi * 8);
        a0.h[1] = *(const v8bf*)(arow0 + k0 + 16 + hi * 8);
        a1.h[0] = *(const v8bf*)(arow1 + k0 + hi * 8);
        a1.h[1] = *(const v8bf*)(arow1 + k0 + 16 + hi * 8);
#pragma unroll
        for (int t = 0; t < 4; ++t) {
            const bf16_t* brow =
                Bt + (long long)(n0 + t * 16 + li) * K + k0 + hi * 8;
            V16U b;
            b.h[0] = *(const v8bf*)(brow);
            b.h[1] = *(const v8bf*)(brow + 16);
            acc[0][t] = wmma_bf16(a0.v, b.v, acc[0][t]);
            acc[1][t] = wmma_bf16(a1.v, b.v, acc[1][t]);
        }
    }

#pragma unroll
    for (int r = 0; r < 2; ++r) {
#pragma unroll
        for (int t = 0; t < 4; ++t) {
            const int col = n0 + t * 16 + li;
#pragma unroll
            for (int j = 0; j < 8; ++j) {
                const long long row = m0 + r * 16 + j + 8 * hi;
                if (OUT_F32)
                    ((float*)Cout)[row * N + col] = acc[r][t][j];
                else
                    ((bf16_t*)Cout)[row * N + col] = f2bf(acc[r][t][j]);
            }
        }
    }
}

// ---------------------------------------------------------------------------
// 4) RoPE + qk RMSNorm + head reshape. One wave handles one (b,s,h) row.
//    qkv bf16 [B,S,3E] -> qn,kn bf16 [B,H,S,D], vT bf16 [B,H,D,S]
// ---------------------------------------------------------------------------
__global__ void rope_norm_reshape_kernel(const bf16_t* __restrict__ qkv,
                                         bf16_t* __restrict__ qn,
                                         bf16_t* __restrict__ kn,
                                         bf16_t* __restrict__ vt) {
    const int lane = threadIdx.x & 31;
    const int wave = threadIdx.x >> 5;
    const int rid = blockIdx.x * 8 + wave;       // 0 .. B*S*H-1
    const int b = rid / (CS * CH);
    const int rem = rid % (CS * CH);
    const int s = rem / CH;
    const int h = rem % CH;

    const long long base = ((long long)(b * CS + s)) * (3 * CE) + (long long)h * CD;

    float qv[4], kv[4], vv[4];
#pragma unroll
    for (int i = 0; i < 4; ++i) {
        const int d = lane * 4 + i;
        qv[i] = bf2f(qkv[base + d]);
        kv[i] = bf2f(qkv[base + CE + d]);
        vv[i] = bf2f(qkv[base + 2 * CE + d]);
    }

    // RoPE on pairs (0,1) and (2,3): pair index = lane*2 + p in [0,64)
#pragma unroll
    for (int p = 0; p < 2; ++p) {
        const int pi = lane * 2 + p;
        // inv_freq = 10000^(-2*pi/128) = exp(-2*pi * ln(10000)/128)
        const float ang =
            (float)s * __expf(-(float)(2 * pi) * (9.210340371976184f / 128.0f));
        const float c = __cosf(ang), sn = __sinf(ang);
        const float qe = qv[2 * p], qo = qv[2 * p + 1];
        qv[2 * p]     = qe * c - qo * sn;
        qv[2 * p + 1] = qe * sn + qo * c;
        const float ke = kv[2 * p], ko = kv[2 * p + 1];
        kv[2 * p]     = ke * c - ko * sn;
        kv[2 * p + 1] = ke * sn + ko * c;
    }

    // RMSNorm over D=128 (4 values per lane, 32 lanes)
    float ssq = 0.0f, ssk = 0.0f;
#pragma unroll
    for (int i = 0; i < 4; ++i) { ssq += qv[i] * qv[i]; ssk += kv[i] * kv[i]; }
#pragma unroll
    for (int off = 16; off > 0; off >>= 1) {
        ssq += __shfl_xor(ssq, off, 32);
        ssk += __shfl_xor(ssk, off, 32);
    }
    const float rq = rsqrtf(ssq / 128.0f + 1e-5f);
    const float rk = rsqrtf(ssk / 128.0f + 1e-5f);

    const long long obase = ((long long)(b * CH + h) * CS + s) * CD;
#pragma unroll
    for (int i = 0; i < 4; ++i) {
        const int d = lane * 4 + i;
        qn[obase + d] = f2bf(qv[i] * rq);
        kn[obase + d] = f2bf(kv[i] * rk);
        vt[((long long)(b * CH + h) * CD + d) * CS + s] = f2bf(vv[i]);
    }
}

// ---------------------------------------------------------------------------
// 5) Flash attention, TRANSPOSED score tiles (S^T = K Q^T, O^T = V^T P^T).
//    K/V tiles staged in LDS via CDNA5 async global->LDS DMA (ASYNCcnt).
//    grid = (S/128, B*H), block = 256 (8 waves), each wave owns 16 q rows.
//    Lane layout: li = q column (one q row per lane), hiL selects k/d half.
//    Q,K: [B,H,S,D] bf16. Vt: [B,H,D,S] bf16. Y: [B,S,E] bf16 (E = h*D+d).
// ---------------------------------------------------------------------------
__global__ __launch_bounds__(256, 1) void flash_attn_kernel(
        const bf16_t* __restrict__ Q,
        const bf16_t* __restrict__ Km,
        const bf16_t* __restrict__ Vt,
        bf16_t* __restrict__ Y) {
    __shared__ __align__(16) bf16_t Ks[32 * 128];   // K tile: 32 k-rows x 128 d (8KB)
    __shared__ __align__(16) bf16_t Vs[128 * 32];   // V tile: 128 d-rows x 32 k (8KB)
    __shared__ __align__(16) bf16_t Ps[8 * 512];    // per-wave P^T tile 16q x 32k (8KB)

    const int tid = threadIdx.x;
    const int lane = tid & 31;
    const int wave = tid >> 5;
    const int li = lane & 15;
    const int hiL = lane >> 4;
    const int bh = blockIdx.y;
    const int b = bh / CH, h = bh % CH;
    const int qr0 = blockIdx.x * 128 + wave * 16;

    const long long hbase = (long long)bh * CS * CD;
    const bf16_t* qbase = Q + hbase;
    const bf16_t* kbase = Km + hbase;
    const bf16_t* vtb = Vt + (long long)bh * CD * CS;
    bf16_t* pw = Ps + wave * 512;

    // Q as persistent B-fragments: lane li = q column (row qr0+li), split-8 d segments
    V16U qf[4];
    {
        const bf16_t* qrow = qbase + (long long)(qr0 + li) * CD + hiL * 8;
#pragma unroll
        for (int dc = 0; dc < 4; ++dc) {
            qf[dc].h[0] = *(const v8bf*)(qrow + dc * 32);
            qf[dc].h[1] = *(const v8bf*)(qrow + dc * 32 + 16);
        }
    }

    v8f O[8];                                       // O^T: [d-chunk][d-row 8*hiL+j] per q col
#pragma unroll
    for (int dc = 0; dc < 8; ++dc) O[dc] = v8f_zero();
    float mrun = -3.0e38f, lrun = 0.0f;             // per-lane (per q row) stats

    const float scale = 0.08838834764831845f;       // 1/sqrt(128)
    const int nKBwave = ((qr0 + 15) >> 5) + 1;      // this wave's causal bound
    const int nKBblk = blockIdx.x * 4 + 4;          // block-wide k-block count

    // cooperative-load indices (uniform per thread)
    const int krow_ld = tid >> 3;                   // 0..31
    const int kseg_ld = (tid & 7) * 16;             // 0..112 halfs
    const int vrow_ld = tid >> 1;                   // 0..127
    const int vseg_ld = (tid & 1) * 16;             // 0 or 16 halfs

#pragma unroll 1
    for (int kb = 0; kb < nKBblk; ++kb) {
        const int n0 = kb * 32;

        __syncthreads();   // previous tile fully consumed
        {   // async DMA K (32x128) and V (128x32) tiles into LDS
            async_copy32(kbase + (long long)(n0 + krow_ld) * CD + kseg_ld,
                         Ks + krow_ld * 128 + kseg_ld);
            async_copy32(vtb + (long long)vrow_ld * CS + n0 + vseg_ld,
                         Vs + vrow_ld * 32 + vseg_ld);
            asm volatile("s_wait_asynccnt 0" ::: "memory");
        }
        __syncthreads();   // tile visible to all waves

        if (kb < nKBwave) {   // wave-uniform: skip compute past causal frontier
            // ---- two 16-k sub-tiles: S^T = K_tile * Q^T ----
#pragma unroll
            for (int t = 0; t < 2; ++t) {
                v8f sacc = v8f_zero();
#pragma unroll
                for (int dc = 0; dc < 4; ++dc) {
                    // A-fragment: lane li = k row (t*16+li), split-8 d segments
                    const bf16_t* kp =
                        Ks + (t * 16 + li) * 128 + dc * 32 + hiL * 8;
                    V16U kf;
                    kf.h[0] = *(const v8bf*)(kp);
                    kf.h[1] = *(const v8bf*)(kp + 16);
                    sacc = wmma_bf16(kf.v, qf[dc].v, sacc);
                }
#pragma unroll
                for (int j = 0; j < 8; ++j) sacc[j] *= scale;

                if (n0 + 31 > qr0) {            // diagonal block: causal mask
                    // k(j) = n0 + t*16 + j + 8*hiL ; q = qr0 + li
                    // mask iff k > q  <=>  j > mj
                    const int mj = qr0 + li - n0 - t * 16 - 8 * hiL;
#pragma unroll
                    for (int j = 0; j < 8; ++j)
                        if (j > mj) sacc[j] = -3.0e38f;
                }

                // per-lane online softmax (one q row per lane)
                float tm = sacc[0];
#pragma unroll
                for (int j = 1; j < 8; ++j) tm = fmaxf(tm, sacc[j]);
                tm = fmaxf(tm, __shfl_xor(tm, 16, 32));   // combine k halves
                const float mnew = fmaxf(mrun, tm);
                const float sc = __expf(mrun - mnew);
                v8bf pv;
                float ps = 0.0f;
#pragma unroll
                for (int j = 0; j < 8; ++j) {
                    const float p = __expf(sacc[j] - mnew);
                    ps += p;
                    pv[j] = f2bf(p);
                }
                ps += __shfl_xor(ps, 16, 32);             // combine k halves
                lrun = lrun * sc + ps;
                mrun = mnew;
#pragma unroll
                for (int dc = 0; dc < 8; ++dc) O[dc] = O[dc] * sc;
                // P^T store: row q=li, k = t*16 + 8*hiL + j  (8 contiguous bf16)
                *(v8bf*)(pw + li * 32 + t * 16 + hiL * 8) = pv;
            }

            asm volatile("s_wait_dscnt 0" ::: "memory");

            // ---- P^T as B-fragment: lane li = q column, split-8 k segments ----
            V16U pf;
            pf.h[0] = *(const v8bf*)(pw + li * 32 + hiL * 8);
            pf.h[1] = *(const v8bf*)(pw + li * 32 + hiL * 8 + 16);

            // ---- O^T += V^T @ P^T over all 8 d-chunks (V^T from LDS as A) ----
#pragma unroll
            for (int dc = 0; dc < 8; ++dc) {
                const bf16_t* vp = Vs + (dc * 16 + li) * 32 + hiL * 8;
                V16U vf;
                vf.h[0] = *(const v8bf*)(vp);
                vf.h[1] = *(const v8bf*)(vp + 16);
                O[dc] = wmma_bf16(vf.v, pf.v, O[dc]);
            }
        }
    }

    // ---- epilogue: normalize, pack 8 bf16 per chunk, one b128 store each ----
    const float inv = 1.0f / lrun;
    bf16_t* yrow = Y + ((long long)(b * CS + qr0 + li)) * CE + h * CD + 8 * hiL;
#pragma unroll
    for (int dc = 0; dc < 8; ++dc) {
        v8bf ov;
#pragma unroll
        for (int j = 0; j < 8; ++j) ov[j] = f2bf(O[dc][j] * inv);
        *(v8bf*)(yrow + dc * 16) = ov;
    }
}

// ---------------------------------------------------------------------------
// Launcher
// ---------------------------------------------------------------------------
extern "C" void kernel_launch(void* const* d_in, const int* in_sizes, int n_in,
                              void* d_out, int out_size, void* d_ws, size_t ws_size,
                              hipStream_t stream) {
    (void)in_sizes; (void)n_in; (void)out_size; (void)ws_size;
    const float* x = (const float*)d_in[0];
    const float* w_qkv = (const float*)d_in[1];
    const float* w_proj = (const float*)d_in[2];
    float* out = (float*)d_out;

    char* ws = (char*)d_ws;
    size_t off = 0;
    bf16_t* xb = (bf16_t*)(ws + off);     off += (size_t)CM * CE * 2;          // 32 MB
    bf16_t* wqkvt = (bf16_t*)(ws + off);  off += (size_t)3 * CE * CE * 2;      // 24 MB
    bf16_t* wprojt = (bf16_t*)(ws + off); off += (size_t)CE * CE * 2;          // 8 MB
    bf16_t* qkv = (bf16_t*)(ws + off);    off += (size_t)CM * 3 * CE * 2;      // 96 MB
    bf16_t* qn = (bf16_t*)(ws + off);     off += (size_t)CM * CE * 2;          // 32 MB
    bf16_t* kn = (bf16_t*)(ws + off);     off += (size_t)CM * CE * 2;          // 32 MB
    bf16_t* vt = (bf16_t*)(ws + off);     off += (size_t)CM * CE * 2;          // 32 MB
    bf16_t* yb = (bf16_t*)(ws + off);     off += (size_t)CM * CE * 2;          // 32 MB

    // 1) x -> bf16
    {
        long long n = (long long)CM * CE;
        int blocks = (int)((n / 4 + 255) / 256);
        cvt_f32_bf16_kernel<<<blocks, 256, 0, stream>>>(x, xb, n);
    }
    // 2) weights -> transposed bf16
    {
        long long n1 = (long long)CE * 3 * CE;
        transpose_cvt_kernel<<<(int)((n1 + 255) / 256), 256, 0, stream>>>(
            w_qkv, wqkvt, CE, 3 * CE);
        long long n2 = (long long)CE * CE;
        transpose_cvt_kernel<<<(int)((n2 + 255) / 256), 256, 0, stream>>>(
            w_proj, wprojt, CE, CE);
    }
    // 3) qkv = x @ w_qkv   (M=8192, N=6144, K=2048)
    gemm_bf16_kernel<false><<<dim3(CM / 64, (3 * CE) / 256), 256, 0, stream>>>(
        xb, wqkvt, qkv, CM, 3 * CE, CE);
    // 4) rope + rmsnorm + head reshape
    rope_norm_reshape_kernel<<<(CB * CS * CH) / 8, 256, 0, stream>>>(
        qkv, qn, kn, vt);
    // 5) causal flash attention
    flash_attn_kernel<<<dim3(CS / 128, CB * CH), 256, 0, stream>>>(
        qn, kn, vt, yb);
    // 6) out = y @ w_proj  (M=8192, N=2048, K=2048), f32 output
    gemm_bf16_kernel<true><<<dim3(CM / 64, CE / 256), 256, 0, stream>>>(
        yb, wprojt, out, CM, CE, CE);
}
